// GGMLLinear_3298534883408
// MI455X (gfx1250) — matrix operand order repaired and checked by reading, not compiled, not run on hardware
//
#include <hip/hip_runtime.h>
#include <stdint.h>

// ---------------------------------------------------------------------------
// Q4_0 GGML linear: out[8192, 11008] = x[8192, 4096] * dequant(W)^T + bias
// Memory-bound (~520 MB min HBM traffic -> ~22 us @ 23.3 TB/s).
//  - f16 WMMA, f32 accumulate, 128x128 workgroup tile, BK=64 (= 2 q4 blocks,
//    keeps every 36B weight group 4B-aligned).
//  - Register double-buffering: next iteration's global loads issued before
//    the WMMA phase so HBM/L2 latency hides behind 16 WMMAs.
//  - Non-temporal output stores: keep the 25 MB quantized weights (re-read by
//    all 64 M-supertiles) resident in the 192 MB L2.
// ---------------------------------------------------------------------------

typedef __attribute__((ext_vector_type(16))) _Float16 v16h;
typedef __attribute__((ext_vector_type(8)))  _Float16 hv8;
typedef __attribute__((ext_vector_type(4)))  _Float16 hv4;
typedef __attribute__((ext_vector_type(8)))  float    v8f;

#define IN_F   4096
#define OUT_F  11008
#define M_TOT  8192
#define BM     128
#define BN     128
#define BK     64
#define KPAD   72                      // BK + 8 halves pad -> 144B row stride, conflict-free
#define WROW_BYTES 2304                // (4096/32) blocks * 18 bytes
#define NIT    (IN_F / BK)             // 64

__device__ __forceinline__ v16h cat8(hv8 a, hv8 b) {
  v16h r;
#pragma unroll
  for (int i = 0; i < 8; ++i) { r[i] = a[i]; r[i + 8] = b[i]; }
  return r;
}

__global__ __launch_bounds__(256) void q4_0_gemm_wmma(
    const float* __restrict__ x,
    const uint8_t* __restrict__ w,
    const float* __restrict__ bias,
    float* __restrict__ out)
{
  __shared__ _Float16 As[BM][KPAD];   // x tile, f16
  __shared__ _Float16 Bs[BN][KPAD];   // dequantized weight tile, f16 (row = N, col = K)

  const int tid  = threadIdx.x;
  const int m0   = blockIdx.y * BM;
  const int n0   = blockIdx.x * BN;

  const int lane  = tid & 31;
  const int wave  = tid >> 5;         // 0..7 (wave32)
  const int waveM = wave >> 1;        // 0..3 -> 32-row slice
  const int waveN = wave & 1;         // 0..1 -> 64-col slice

  const int l16   = lane & 15;
  const int lhalf = lane >> 4;        // 0 or 1

  // Fragment addressing per CDNA5 ISA §7.12.2:
  //  A (16x32 f16): lane<16 holds K 0-7 & 16-23 of row l; lane>=16 holds K 8-15 & 24-31
  //  B (32x16 f16): lane<16 holds K 0-7 & 16-23 of col l; lane>=16 holds K 8-15 & 24-31
  const int aRow  = waveM * 32 + l16;
  const int aKh   = lhalf * 8;
  const int bCol  = waveN * 64 + l16;
  const int bKo   = lhalf * 8;

  // ---- staging helpers (register double buffer) ----
  float4   abuf[8];                   // 128x64 f32 tile / 256 threads = 8 float4 each
  uint4    wq0, wq1;                  // 2 q4_0 blocks = 36B (threads 0..127, one N-row each)
  uint32_t wq2;

  auto load_a = [&](int it, float4 (&buf)[8]) {
    const int k0 = it * BK;
#pragma unroll
    for (int i = 0; i < 8; ++i) {
      const int idx = i * 256 + tid;  // float4 index within tile (2048 total)
      const int row = idx >> 4;       // 16 float4 per row
      const int c4  = idx & 15;
      buf[i] = *(const float4*)(x + (size_t)(m0 + row) * IN_F + k0 + c4 * 4);
    }
  };
  auto load_w = [&](int it) {
    const uint8_t* gp = w + (size_t)(n0 + tid) * WROW_BYTES + (size_t)it * 36;
    wq0 = *(const uint4*)(gp);
    wq1 = *(const uint4*)(gp + 16);
    wq2 = *(const uint32_t*)(gp + 32);
  };
  auto store_a = [&](const float4 (&buf)[8]) {
#pragma unroll
    for (int i = 0; i < 8; ++i) {
      const int idx = i * 256 + tid;
      const int row = idx >> 4;
      const int c4  = idx & 15;
      hv4 h;
      h[0] = (_Float16)buf[i].x; h[1] = (_Float16)buf[i].y;
      h[2] = (_Float16)buf[i].z; h[3] = (_Float16)buf[i].w;
      *(hv4*)&As[row][c4 * 4] = h;
    }
  };
  auto store_w = [&]() {
    uint32_t w9[9] = {wq0.x, wq0.y, wq0.z, wq0.w, wq1.x, wq1.y, wq1.z, wq1.w, wq2};
    auto getb = [&](int i) -> uint32_t { return (w9[i >> 2] >> ((i & 3) * 8)) & 0xffu; };
#pragma unroll
    for (int blk = 0; blk < 2; ++blk) {
      const int base = blk * 18;
      const uint16_t dbits = (uint16_t)(getb(base) | (getb(base + 1) << 8));
      _Float16 d;
      __builtin_memcpy(&d, &dbits, 2);
      hv8 o[4];
#pragma unroll
      for (int j = 0; j < 16; ++j) {
        const uint32_t q = getb(base + 2 + j);
        o[(j >> 3)][j & 7]     = d * (_Float16)((int)(q & 15u) - 8);  // k = j
        o[2 + (j >> 3)][j & 7] = d * (_Float16)((int)(q >> 4) - 8);  // k = 16 + j
      }
#pragma unroll
      for (int g = 0; g < 4; ++g)
        *(hv8*)&Bs[tid][blk * 32 + g * 8] = o[g];
    }
  };

  // ---- prologue: fetch iteration 0 ----
  load_a(0, abuf);
  if (tid < 128) load_w(0);

  v8f acc[2][4] = {};                 // 2 M-subtiles x 4 N-subtiles of 16x16

  for (int it = 0; it < NIT; ++it) {
    // regs -> LDS (waits on the in-flight global loads here)
    store_a(abuf);
    if (tid < 128) store_w();
    __syncthreads();

    // issue next iteration's global loads; latency hides behind the WMMAs
    if (it + 1 < NIT) {
      load_a(it + 1, abuf);
      if (tid < 128) load_w(it + 1);
    }

#pragma unroll
    for (int ks = 0; ks < BK; ks += 32) {
      v16h afrag[2];
#pragma unroll
      for (int sm = 0; sm < 2; ++sm) {
        const hv8 lo = *(const hv8*)&As[aRow + sm * 16][ks + aKh];
        const hv8 hi = *(const hv8*)&As[aRow + sm * 16][ks + 16 + aKh];
        afrag[sm] = cat8(lo, hi);
      }
      v16h bfrag[4];
#pragma unroll
      for (int sn = 0; sn < 4; ++sn) {
        const hv8 lo = *(const hv8*)&Bs[bCol + sn * 16][ks + bKo];
        const hv8 hi = *(const hv8*)&Bs[bCol + sn * 16][ks + bKo + 16];
        bfrag[sn] = cat8(lo, hi);
      }
#pragma unroll
      for (int sm = 0; sm < 2; ++sm)
#pragma unroll
        for (int sn = 0; sn < 4; ++sn)
          acc[sm][sn] = __builtin_amdgcn_wmma_f32_16x16x32_f16(
              false, afrag[sm], false, bfrag[sn],
              (short)0, acc[sm][sn], false, false);
    }

    __syncthreads();
  }

  // ---- epilogue: C/D layout (lane<16: M=v, N=lane; lane>=16: M=v+8, N=lane-16) ----
  // Non-temporal stores: don't let the 360 MB output stream evict the
  // L2-resident quantized weights / active x slice.
#pragma unroll
  for (int sn = 0; sn < 4; ++sn) {
    const int n = n0 + waveN * 64 + sn * 16 + l16;
    const float bval = bias[n];
#pragma unroll
    for (int sm = 0; sm < 2; ++sm) {
      const int mBase = m0 + waveM * 32 + sm * 16 + lhalf * 8;
#pragma unroll
      for (int v = 0; v < 8; ++v) {
        __builtin_nontemporal_store(acc[sm][sn][v] + bval,
                                    &out[(size_t)(mBase + v) * OUT_F + n]);
      }
    }
  }
}

extern "C" void kernel_launch(void* const* d_in, const int* in_sizes, int n_in,
                              void* d_out, int out_size, void* d_ws, size_t ws_size,
                              hipStream_t stream) {
  const float*   xx = (const float*)d_in[0];
  const uint8_t* ww = (const uint8_t*)d_in[1];
  const float*   bb = (const float*)d_in[2];
  float*         oo = (float*)d_out;
  (void)in_sizes; (void)n_in; (void)out_size; (void)d_ws; (void)ws_size;

  dim3 grid(OUT_F / BN, M_TOT / BM);   // (86, 64)
  q4_0_gemm_wmma<<<grid, 256, 0, stream>>>(xx, ww, bb, oo);
}
